// SplitDelta_35588099014894
// MI455X (gfx1250) — matrix operand order repaired and checked by reading, not compiled, not run on hardware
//
#include <hip/hip_runtime.h>

#define HIDDEN 64
#define H2     128
#define HALF   32
#define VOCABN 64
#define BB     256
#define LL     2048

typedef _Float16 v16h __attribute__((ext_vector_type(16)));
typedef float    v8f  __attribute__((ext_vector_type(8)));

// ---- WMMA fragment builders (CDNA5 16-bit layouts, wave32) ----
// A 16x32 f16: lane -> M = lane&15, half = lane>>4; packed p -> K = (p>>3)*16 + half*8 + (p&7)
__device__ __forceinline__ v16h load_afrag(const _Float16* X, int stride, int k0, int lane) {
  int M = lane & 15, hh = lane >> 4;
  v16h f;
#pragma unroll
  for (int p = 0; p < 16; ++p) {
    int K = k0 + ((p >> 3) << 4) + hh * 8 + (p & 7);
    f[p] = X[M * stride + K];
  }
  return f;
}
// B 32x16 f16: lane -> N = lane&15, kblk = lane>>4; packed p -> K = (p>>3)*16 + kblk*8 + (p&7)
// Wt stored [N][K] row-major so each lane's 16 halves are two contiguous 8-half runs (2x ds_load_b128)
__device__ __forceinline__ v16h load_bfrag(const _Float16* Wt, int stride, int n0, int k0, int lane) {
  int N = lane & 15, kb = lane >> 4;
  const _Float16* src = Wt + (n0 + N) * stride + k0 + kb * 8;
  v16h f;
#pragma unroll
  for (int p = 0; p < 16; ++p) f[p] = src[((p >> 3) << 4) + (p & 7)];
  return f;
}

// Wave-uniform broadcast via SALU readlane (co-executes with VALU FMAs)
__device__ __forceinline__ float bcast(float v, int srcLane) {
  return __builtin_bit_cast(float, __builtin_amdgcn_readlane(__builtin_bit_cast(int, v), srcLane));
}

// ================= Phase 1: encoder (embed->FF->LN->hs/he), WMMA f16 =================
__global__ __launch_bounds__(256) void sd_encode_kernel(
    const int* __restrict__ seq, const float* __restrict__ embed,
    const float* __restrict__ W1, const float* __restrict__ b1,
    const float* __restrict__ W2, const float* __restrict__ b2,
    const float* __restrict__ ln_g, const float* __restrict__ ln_b,
    const float* __restrict__ Ws, const float* __restrict__ bs,
    const float* __restrict__ We, const float* __restrict__ be,
    _Float16* __restrict__ hsb, _Float16* __restrict__ heb) {
  __shared__ _Float16 sW1[H2 * HIDDEN];     // W1 [128][64] = W1.T as [N][K]
  __shared__ _Float16 sW2[HIDDEN * H2];     // W2 [64][128] = W2.T as [N][K]
  __shared__ _Float16 sWs[HALF * HIDDEN];
  __shared__ _Float16 sWe[HALF * HIDDEN];
  __shared__ float    sEmb[VOCABN * HIDDEN];
  __shared__ _Float16 sT[8][16 * H2];       // per-wave transpose staging
  __shared__ int      sSeq[8][16];
  __shared__ float sB1[H2], sB2[HIDDEN], sG[HIDDEN], sBt[HIDDEN], sBs[HALF], sBe[HALF];

  const int tid = threadIdx.x;
  const int w = tid >> 5;
  const int lane = tid & 31;

  for (int i = tid; i < H2 * HIDDEN; i += 256) sW1[i] = (_Float16)W1[i];
  for (int i = tid; i < HIDDEN * H2; i += 256) sW2[i] = (_Float16)W2[i];
  for (int i = tid; i < HALF * HIDDEN; i += 256) { sWs[i] = (_Float16)Ws[i]; sWe[i] = (_Float16)We[i]; }
  for (int i = tid; i < VOCABN * HIDDEN; i += 256) sEmb[i] = embed[i];
  if (tid < H2) sB1[tid] = b1[tid];
  if (tid < HIDDEN) { sB2[tid] = b2[tid]; sG[tid] = ln_g[tid]; sBt[tid] = ln_b[tid]; }
  if (tid < HALF) { sBs[tid] = bs[tid]; sBe[tid] = be[tid]; }

  const int tileTok = (blockIdx.x * 8 + w) * 16;
  if (lane < 16) sSeq[w][lane] = seq[tileTok + lane];
  __syncthreads();

  // ---- A fragments of h0 via embed gather ----
  const int Ma = lane & 15;
  const int hh = lane >> 4;
  const int erow = sSeq[w][Ma];
  v16h a0, a1;
#pragma unroll
  for (int p = 0; p < 16; ++p) {
    int K = ((p >> 3) << 4) + hh * 8 + (p & 7);
    a0[p] = (_Float16)sEmb[erow * HIDDEN + K];
    a1[p] = (_Float16)sEmb[erow * HIDDEN + 32 + K];
  }

  // ---- GEMM1: u = relu(h0 @ W1.T + b1) -> sT[w] [16][128] f16 ----
  const int Nc = lane & 15;
  const int hc = lane >> 4;
#pragma unroll
  for (int nt = 0; nt < 8; ++nt) {
    v8f c = {};
    c = __builtin_amdgcn_wmma_f32_16x16x32_f16(false, a0, false, load_bfrag(sW1, HIDDEN, nt * 16, 0, lane), (short)0, c, false, false);
    c = __builtin_amdgcn_wmma_f32_16x16x32_f16(false, a1, false, load_bfrag(sW1, HIDDEN, nt * 16, 32, lane), (short)0, c, false, false);
    float bias = sB1[nt * 16 + Nc];
#pragma unroll
    for (int r = 0; r < 8; ++r) {
      float v = c[r] + bias;
      v = v > 0.f ? v : 0.f;
      sT[w][(r + 8 * hc) * H2 + nt * 16 + Nc] = (_Float16)v;
    }
  }
  __syncthreads();

  // ---- GEMM2: x = u @ W2.T + b2 + h0 ----
  v16h a2[4];
#pragma unroll
  for (int kb = 0; kb < 4; ++kb) a2[kb] = load_afrag(sT[w], H2, kb * 32, lane);

  v8f x[4];
#pragma unroll
  for (int nt = 0; nt < 4; ++nt) {
    v8f c = {};
#pragma unroll
    for (int kb = 0; kb < 4; ++kb)
      c = __builtin_amdgcn_wmma_f32_16x16x32_f16(false, a2[kb], false, load_bfrag(sW2, H2, nt * 16, kb * 32, lane), (short)0, c, false, false);
    int N = nt * 16 + Nc;
    float bias = sB2[N];
#pragma unroll
    for (int r = 0; r < 8; ++r) {
      int Mr = r + 8 * hc;
      c[r] += bias + sEmb[sSeq[w][Mr] * HIDDEN + N];  // residual h0
    }
    x[nt] = c;
  }

  // ---- LayerNorm over hidden (16-lane half-wave reduction per token row) ----
  float mu[8], ri[8];
#pragma unroll
  for (int r = 0; r < 8; ++r) {
    float s1 = x[0][r] + x[1][r] + x[2][r] + x[3][r];
    float s2 = x[0][r]*x[0][r] + x[1][r]*x[1][r] + x[2][r]*x[2][r] + x[3][r]*x[3][r];
#pragma unroll
    for (int m = 1; m <= 8; m <<= 1) {
      s1 += __shfl_xor(s1, m, 32);
      s2 += __shfl_xor(s2, m, 32);
    }
    float mean = s1 * (1.0f / 64.0f);
    float var  = s2 * (1.0f / 64.0f) - mean * mean;
    mu[r] = mean;
    ri[r] = rsqrtf(var + 1e-5f);
  }
  __syncthreads();
#pragma unroll
  for (int nt = 0; nt < 4; ++nt) {
    int N = nt * 16 + Nc;
    float g = sG[N], bt = sBt[N];
#pragma unroll
    for (int r = 0; r < 8; ++r) {
      float v = (x[nt][r] - mu[r]) * ri[r] * g + bt;
      sT[w][(r + 8 * hc) * HIDDEN + N] = (_Float16)v;
    }
  }
  __syncthreads();

  // ---- GEMM3: hs = h @ Ws.T + bs ; he = h @ We.T + be -> global f16 ----
  v16h a3lo = load_afrag(sT[w], HIDDEN, 0, lane);
  v16h a3hi = load_afrag(sT[w], HIDDEN, 32, lane);
#pragma unroll
  for (int nt = 0; nt < 2; ++nt) {
    v8f cs = {}, ce = {};
    cs = __builtin_amdgcn_wmma_f32_16x16x32_f16(false, a3lo, false, load_bfrag(sWs, HIDDEN, nt * 16, 0, lane), (short)0, cs, false, false);
    cs = __builtin_amdgcn_wmma_f32_16x16x32_f16(false, a3hi, false, load_bfrag(sWs, HIDDEN, nt * 16, 32, lane), (short)0, cs, false, false);
    ce = __builtin_amdgcn_wmma_f32_16x16x32_f16(false, a3lo, false, load_bfrag(sWe, HIDDEN, nt * 16, 0, lane), (short)0, ce, false, false);
    ce = __builtin_amdgcn_wmma_f32_16x16x32_f16(false, a3hi, false, load_bfrag(sWe, HIDDEN, nt * 16, 32, lane), (short)0, ce, false, false);
    int N = nt * 16 + Nc;
    float bbs = sBs[N], bbe = sBe[N];
#pragma unroll
    for (int r = 0; r < 8; ++r) {
      int tok = tileTok + r + 8 * hc;
      hsb[(size_t)tok * HALF + N] = (_Float16)(cs[r] + bbs);
      heb[(size_t)tok * HALF + N] = (_Float16)(ce[r] + bbe);
    }
  }
}

// ================= Phase 2: delta-rule scan =================
// One wave per (batch, memory); lane i owns row i of the 32x32 state in VGPRs.
// k_j broadcast via SALU readlane (uniform SGPR operand to the FMAs);
// vp reduced with 4 partial accumulators (dependent chain 32 -> ~10);
// 4-deep software pipeline on the streamed key component (~600 cycles of cover).
__global__ __launch_bounds__(256) void sd_scan_kernel(const _Float16* __restrict__ hsb,
                                                      const _Float16* __restrict__ heb,
                                                      float* __restrict__ rbuf) {
  const int gid = blockIdx.x * blockDim.x + threadIdx.x;
  const int wid = gid >> 5;
  const int lane = threadIdx.x & 31;
  const int b = wid >> 1;
  const int mem = wid & 1;
  const _Float16* src = (mem ? heb : hsb) + (size_t)b * LL * HALF;

  float Mr[32];
#pragma unroll
  for (int j = 0; j < 32; ++j) Mr[j] = 0.f;

  const float invL = 1.0f / (float)LL;

  // prime 4-deep pipeline with steps t = 0..3 (component `lane`)
  float pipe[4];
#pragma unroll
  for (int i = 0; i < 4; ++i) pipe[i] = (float)src[(size_t)i * HALF + lane];

  for (int t = 0; t < LL - 1; ++t) {
    float kv = pipe[t & 3];
    int tn = t + 4;
    if (tn < LL) pipe[t & 3] = (float)src[(size_t)tn * HALF + lane];

    // l2norm (eps matches F.normalize)
    float ss = kv * kv;
#pragma unroll
    for (int m = 16; m >= 1; m >>= 1) ss += __shfl_xor(ss, m, 32);
    float inv = 1.0f / fmaxf(sqrtf(ss), 1e-12f);
    float k = kv * inv;

    // uniform broadcast of all 32 key components into SGPRs
    float kb[32];
#pragma unroll
    for (int j = 0; j < 32; ++j) kb[j] = bcast(k, j);

    // vp = M k with 4 partial chains
    float vp0 = 0.f, vp1 = 0.f, vp2 = 0.f, vp3 = 0.f;
#pragma unroll
    for (int j = 0; j < 8; ++j) {
      vp0 = fmaf(Mr[j],      kb[j],      vp0);
      vp1 = fmaf(Mr[j + 8],  kb[j + 8],  vp1);
      vp2 = fmaf(Mr[j + 16], kb[j + 16], vp2);
      vp3 = fmaf(Mr[j + 24], kb[j + 24], vp3);
    }
    float vp = (vp0 + vp1) + (vp2 + vp3);

    float sc = mem ? (float)(t + 1) * invL : 1.0f;  // (t+1)/L only on M_e
    float d = (kv - vp) * sc;
#pragma unroll
    for (int j = 0; j < 32; ++j) Mr[j] = fmaf(d, kb[j], Mr[j]);
  }

  // query: step t = L-1 sits in pipe[(LL-1) & 3] (loaded at t = LL-5, never overwritten)
  float kv = pipe[(LL - 1) & 3];
  float ss = kv * kv;
#pragma unroll
  for (int m = 16; m >= 1; m >>= 1) ss += __shfl_xor(ss, m, 32);
  float inv = 1.0f / fmaxf(sqrtf(ss), 1e-12f);
  float q = kv * inv;
  float r0 = 0.f, r1 = 0.f, r2 = 0.f, r3 = 0.f;
#pragma unroll
  for (int j = 0; j < 8; ++j) {
    r0 = fmaf(Mr[j],      bcast(q, j),      r0);
    r1 = fmaf(Mr[j + 8],  bcast(q, j + 8),  r1);
    r2 = fmaf(Mr[j + 16], bcast(q, j + 16), r2);
    r3 = fmaf(Mr[j + 24], bcast(q, j + 24), r3);
  }
  rbuf[b * 64 + mem * HALF + lane] = (r0 + r1) + (r2 + r3);
}

// ================= Phase 3: r @ Wrp.T + brp -> @ Wo.T + bo =================
__global__ void sd_head_kernel(const float* __restrict__ rbuf,
                               const float* __restrict__ Wrp, const float* __restrict__ brp,
                               const float* __restrict__ Wo, const float* __restrict__ bo,
                               float* __restrict__ out) {
  __shared__ float rloc[64];
  __shared__ float tmp[64];
  const int b = blockIdx.x;
  const int t = threadIdx.x;  // 64 threads
  rloc[t] = rbuf[b * 64 + t];
  __syncthreads();
  float acc = brp[t];
#pragma unroll 8
  for (int j = 0; j < 64; ++j) acc += rloc[j] * Wrp[t * 64 + j];
  tmp[t] = acc;
  __syncthreads();
  float o = bo[t];
#pragma unroll 8
  for (int k = 0; k < 64; ++k) o += tmp[k] * Wo[t * 64 + k];
  out[b * 64 + t] = o;
}

extern "C" void kernel_launch(void* const* d_in, const int* in_sizes, int n_in,
                              void* d_out, int out_size, void* d_ws, size_t ws_size,
                              hipStream_t stream) {
  const int*   seq   = (const int*)d_in[0];
  const float* embed = (const float*)d_in[1];
  const float* W1    = (const float*)d_in[2];
  const float* b1    = (const float*)d_in[3];
  const float* W2    = (const float*)d_in[4];
  const float* b2    = (const float*)d_in[5];
  const float* ln_g  = (const float*)d_in[6];
  const float* ln_b  = (const float*)d_in[7];
  const float* Ws    = (const float*)d_in[8];
  const float* bs    = (const float*)d_in[9];
  const float* We    = (const float*)d_in[10];
  const float* be    = (const float*)d_in[11];
  const float* Wrp   = (const float*)d_in[12];
  const float* brp   = (const float*)d_in[13];
  const float* Wo    = (const float*)d_in[14];
  const float* bo    = (const float*)d_in[15];
  float* out = (float*)d_out;

  _Float16* hsb = (_Float16*)d_ws;                        // [B][L][32] f16
  _Float16* heb = hsb + (size_t)BB * LL * HALF;           // [B][L][32] f16
  float* rbuf = (float*)(heb + (size_t)BB * LL * HALF);   // [B][64] f32

  sd_encode_kernel<<<(BB * LL) / 128, 256, 0, stream>>>(seq, embed, W1, b1, W2, b2,
                                                        ln_g, ln_b, Ws, bs, We, be, hsb, heb);
  sd_scan_kernel<<<(BB * 2 * 32) / 256, 256, 0, stream>>>(hsb, heb, rbuf);
  sd_head_kernel<<<BB, 64, 0, stream>>>(rbuf, Wrp, brp, Wo, bo, out);
}
